// FusionModel_68281390071948
// MI455X (gfx1250) — matrix operand order repaired and checked by reading, not compiled, not run on hardware
//
#include <hip/hip_runtime.h>

// ---------------------------------------------------------------------------
// CDNA5 / gfx1250 implementation. All 128/64-wide GEMMs run on
// v_wmma_f32_16x16x32_bf16 (fp32 accumulate). Small-K layers and the M=8
// heads run on VALU. Scatter (seg_mean over arbitrary edge indices) uses
// global_atomic_add_f32.
// ---------------------------------------------------------------------------

typedef __bf16 bf16_t;
typedef __attribute__((ext_vector_type(16))) bf16_t          v16bf;
typedef __attribute__((ext_vector_type(8)))  float           v8f;
typedef __attribute__((ext_vector_type(8)))  unsigned short  v8us;

#define DEVFN __device__ __forceinline__

DEVFN unsigned short f2bf_bits(float f) {               // RNE f32 -> bf16 bits
  union { float f; unsigned u; } v; v.f = f;
  unsigned u = v.u;
  return (unsigned short)((u + 0x7FFFu + ((u >> 16) & 1u)) >> 16);
}
DEVFN float bf2f(unsigned short s) {
  union { unsigned u; float f; } v; v.u = ((unsigned)s) << 16; return v.f;
}
DEVFN bf16_t bits2bf(unsigned short s) {
  union { unsigned short u; bf16_t b; } v; v.u = s; return v.b;
}

// ---------------- model constants ----------------
#define BATCH 8
#define PP    2048
#define KNN   16
#define NTOT  16384           // BATCH*PP
#define ECLOUD 262144         // NTOT*KNN
#define HC    128
#define HT    64
#define SP    20480
#define SE    16384
#define SF    8192

// ---------------------------------------------------------------------------
// KNN: one thread per destination point; keeps 16 smallest d2 (self excluded).
// Also emits edge-encoder inputs [dx,dy,dz,dist] (delta = xyz[dst]-xyz[src]).
// ---------------------------------------------------------------------------
__global__ void knn_kernel(const float* __restrict__ cloud,
                           int* __restrict__ src_idx,
                           float* __restrict__ ef4) {
  int n = blockIdx.x * blockDim.x + threadIdx.x;
  if (n >= NTOT) return;
  int b = n / PP, p = n - b * PP;
  const float* cb = cloud + (size_t)b * PP * 7;
  float x0 = cb[p*7+0], y0 = cb[p*7+1], z0 = cb[p*7+2];
  float bd[KNN]; int bi[KNN];
#pragma unroll
  for (int i = 0; i < KNN; ++i) { bd[i] = 3.0e38f; bi[i] = 0; }
  for (int q = 0; q < PP; ++q) {
    if (q == p) continue;
    float dx = cb[q*7+0]-x0, dy = cb[q*7+1]-y0, dz = cb[q*7+2]-z0;
    float d = dx*dx + dy*dy + dz*dz;
    if (d < bd[KNN-1]) {
      int j = KNN-1;
      while (j > 0 && bd[j-1] > d) { bd[j] = bd[j-1]; bi[j] = bi[j-1]; --j; }
      bd[j] = d; bi[j] = q;
    }
  }
  for (int k = 0; k < KNN; ++k) {
    int q = bi[k];
    src_idx[(size_t)n*KNN + k] = b*PP + q;
    float dx = x0 - cb[q*7+0], dy = y0 - cb[q*7+1], dz = z0 - cb[q*7+2];
    float dist = sqrtf(dx*dx + dy*dy + dz*dz);
    float* o = ef4 + ((size_t)n*KNN + k) * 4;
    o[0] = dx; o[1] = dy; o[2] = dz; o[3] = dist;
  }
}

// ---------------------------------------------------------------------------
// Generic WMMA GEMM: C[M x Nd] (+)= act(A[M x Kd] @ W + bias)
// Wt is the bf16 TRANSPOSED weight [Nd x Kd]. One wave per 16x16 C tile.
// A lane layout (16-bit A 16x32, ISA 7.12.2): half 0 -> K {k0..k0+7, k0+16..23},
// half 1 -> K {k0+8..15, k0+24..31}. B layout: same K split as A,
// col = lane%16.
// ---------------------------------------------------------------------------
template<bool IN_BF16, bool OUT_BF16, bool RELU, bool ACCUM, bool ADD2>
__global__ void wmma_gemm(const void* __restrict__ Ap,
                          const void* __restrict__ A2p,
                          const unsigned short* __restrict__ Wt,
                          const float* __restrict__ bias,
                          void* __restrict__ Cp,
                          int Kd, int Nd) {
  int tm = blockIdx.x, tn = blockIdx.y;
  int lane = threadIdx.x & 31;
  int half = lane >> 4, l16 = lane & 15;
  int row = tm * 16 + l16;
  int col = tn * 16 + l16;
  v8f acc = {};
  for (int k0 = 0; k0 < Kd; k0 += 32) {
    v16bf a, bv;
    int ka = k0 + half * 8;
    if (IN_BF16) {
      const unsigned short* Ab = (const unsigned short*)Ap + (size_t)row * Kd;
      v8us c1 = *(const v8us*)(Ab + ka);
      v8us c2 = *(const v8us*)(Ab + ka + 16);
#pragma unroll
      for (int i = 0; i < 8; ++i) { a[i] = bits2bf(c1[i]); a[i+8] = bits2bf(c2[i]); }
    } else {
      const float* Af  = (const float*)Ap + (size_t)row * Kd;
      const float* A2f = ADD2 ? ((const float*)A2p + (size_t)row * Kd) : (const float*)0;
#pragma unroll
      for (int i = 0; i < 8; ++i) {
        float f1 = Af[ka + i];      if (ADD2) f1 += A2f[ka + i];
        float f2 = Af[ka + 16 + i]; if (ADD2) f2 += A2f[ka + 16 + i];
        a[i]   = bits2bf(f2bf_bits(f1));
        a[i+8] = bits2bf(f2bf_bits(f2));
      }
    }
    {
      const unsigned short* Wr = Wt + (size_t)col * Kd + k0 + half * 8;
      v8us d1 = *(const v8us*)(Wr);
      v8us d2 = *(const v8us*)(Wr + 16);
#pragma unroll
      for (int i = 0; i < 8; ++i) { bv[i] = bits2bf(d1[i]); bv[i+8] = bits2bf(d2[i]); }
    }
    acc = __builtin_amdgcn_wmma_f32_16x16x32_bf16(false, a, false, bv,
                                                  (short)0, acc, false, false);
  }
  float bi = bias ? bias[col] : 0.0f;
#pragma unroll
  for (int v = 0; v < 8; ++v) {
    int r = tm * 16 + half * 8 + v;           // C/D layout: lane half picks M block
    size_t idx = (size_t)r * Nd + col;
    float val = acc[v] + bi;
    if (RELU) val = fmaxf(val, 0.0f);
    if (OUT_BF16) {
      ((unsigned short*)Cp)[idx] = f2bf_bits(val);
    } else {
      float* Cf = (float*)Cp;
      if (ACCUM) val += Cf[idx];
      Cf[idx] = val;
    }
  }
}

// ---------------------------------------------------------------------------
// Small dense layer (VALU): Y[row][j] = act(X[row] . W[:,j] + b[j])
// blockDim = Nd, grid = M. Row of X staged through LDS.
// ---------------------------------------------------------------------------
template<bool RELU, bool OUT_BF16>
__global__ void dense_small(const float* __restrict__ X,
                            const float* __restrict__ W,
                            const float* __restrict__ bias,
                            void* __restrict__ Y, int Kd, int Nd) {
  extern __shared__ float xs[];
  int row = blockIdx.x, j = threadIdx.x;
  for (int k = j; k < Kd; k += Nd) xs[k] = X[(size_t)row * Kd + k];
  __syncthreads();
  float acc = bias[j];
  for (int k = 0; k < Kd; ++k) acc += xs[k] * W[(size_t)k * Nd + j];
  if (RELU) acc = fmaxf(acc, 0.0f);
  size_t idx = (size_t)row * Nd + j;
  if (OUT_BF16) ((unsigned short*)Y)[idx] = f2bf_bits(acc);
  else          ((float*)Y)[idx] = acc;
}

// Transpose + convert fp32 W[Kd x Nd] -> bf16 Wt[Nd x Kd]
__global__ void transpose_bf16(const float* __restrict__ W,
                               unsigned short* __restrict__ Wt, int Kd, int Nd) {
  int i = blockIdx.x * blockDim.x + threadIdx.x;
  if (i >= Kd * Nd) return;
  int k = i / Nd, n = i - k * Nd;
  Wt[(size_t)n * Kd + k] = f2bf_bits(W[(size_t)k * Nd + n]);
}

// GINE aggregation: agg[n][c] = sum_k relu(h[src[n,k]][c] + e_attr[n,k][c])
// dst indices are contiguous (each node owns exactly KNN consecutive edges).
__global__ void gine_agg_kernel(const float* __restrict__ h,
                                const unsigned short* __restrict__ e_attr,
                                const int* __restrict__ src_idx,
                                float* __restrict__ agg) {
  int n = blockIdx.x, c = threadIdx.x;    // blockDim = HC
  __shared__ int s[KNN];
  if (c < KNN) s[c] = src_idx[(size_t)n * KNN + c];
  __syncthreads();
  float a = 0.0f;
#pragma unroll
  for (int k = 0; k < KNN; ++k) {
    float m = h[(size_t)s[k] * HC + c] + bf2f(e_attr[((size_t)n * KNN + k) * HC + c]);
    a += fmaxf(m, 0.0f);
  }
  agg[(size_t)n * HC + c] = a;
}

// y = LayerNorm(x + (RELU_T ? relu(t) : t)) written back into x. blockDim = C.
template<bool RELU_T>
__global__ void ln_add_kernel(float* __restrict__ x, const float* __restrict__ t,
                              const float* __restrict__ g, const float* __restrict__ b,
                              int C) {
  extern __shared__ float sh[];            // 2*C floats
  int row = blockIdx.x, c = threadIdx.x;
  size_t idx = (size_t)row * C + c;
  float tv = t[idx]; if (RELU_T) tv = fmaxf(tv, 0.0f);
  float v = x[idx] + tv;
  sh[c] = v; sh[C + c] = v * v;
  __syncthreads();
  for (int s = C >> 1; s > 0; s >>= 1) {
    if (c < s) { sh[c] += sh[c + s]; sh[C + c] += sh[C + c + s]; }
    __syncthreads();
  }
  float m   = sh[0] / C;
  float var = sh[C] / C - m * m;
  x[idx] = (v - m) * rsqrtf(var + 1e-5f) * g[c] + b[c];
}

__global__ void zero_kernel(float* p, int n) {
  int i = blockIdx.x * blockDim.x + threadIdx.x;
  if (i < n) p[i] = 0.0f;
}

// Scatter-add for seg_mean over arbitrary edge lists. ei = [src[E] | dst[E]].
__global__ void scatter_add_kernel(const float* __restrict__ xs,
                                   const int* __restrict__ ei, int E,
                                   float* __restrict__ accum, float* __restrict__ cnt) {
  int e = blockIdx.x, c = threadIdx.x;     // blockDim = HT
  int s = ei[e], d = ei[E + e];
  atomicAdd(&accum[(size_t)d * HT + c], xs[(size_t)s * HT + c]);
  if (c == 0) atomicAdd(&cnt[d], 1.0f);
}

__global__ void divmean_kernel(float* __restrict__ accum, const float* __restrict__ cnt,
                               int n) {                     // n = rows*HT
  int i = blockIdx.x * blockDim.x + threadIdx.x;
  if (i >= n) return;
  accum[i] /= fmaxf(cnt[i / HT], 1.0f);
}

// Mean over contiguous batch segments. blockDim = C, grid = BATCH.
__global__ void segmean_kernel(const float* __restrict__ x, float* __restrict__ out,
                               int rows_per_b, int C) {
  int b = blockIdx.x, c = threadIdx.x;
  const float* p = x + (size_t)b * rows_per_b * C + c;
  float s = 0.0f;
  for (int r = 0; r < rows_per_b; ++r) s += p[(size_t)r * C];
  out[(size_t)b * C + c] = s / (float)rows_per_b;
}

// head_in[b] = [cloud_pool(128) | g_point(64) | g_edge(64) | g_face(64) | gm(64)]
__global__ void build_head_kernel(const float* cp, const float* gp, const float* ge,
                                  const float* gf, const float* gm, float* out) {
  int b = blockIdx.x, j = threadIdx.x;     // blockDim = 384
  float v;
  if      (j < 128) v = cp[b*128 + j];
  else if (j < 192) v = gp[b*64 + (j-128)];
  else if (j < 256) v = ge[b*64 + (j-192)];
  else if (j < 320) v = gf[b*64 + (j-256)];
  else              v = gm[b*64 + (j-320)];
  out[(size_t)b*384 + j] = v;
}

// ---------------------------------------------------------------------------
// Host orchestration
// ---------------------------------------------------------------------------
extern "C" void kernel_launch(void* const* d_in, const int* in_sizes, int n_in,
                              void* d_out, int out_size, void* d_ws, size_t ws_size,
                              hipStream_t stream) {
  (void)in_sizes; (void)out_size; (void)ws_size;
  if (n_in < 145) return;   // expects 17 tensors + 128 param leaves (pytree order)

  auto F = [&](int i) { return (const float*)d_in[i]; };
  auto I = [&](int i) { return (const int*)d_in[i]; };

  // ---- param leaf indices (JAX pytree: dict keys sorted alphabetically) ----
  const int P_CN   = 17;   // cloud_norms: 6 x (g,b)
  const int P_DH   = 29;   // drop_head: W0,b0,W1,b1,W2,b2
  const int P_EEC  = 35;   // edge_enc_cloud: W0,b0,W1,b1
  const int P_EET  = 39;   // edge_enc_topo
  const int P_FE   = 43;   // face_enc
  const int P_GINE = 47;   // 6 x (W0,b0,W1,b1)
  const int P_GM   = 71;   // global_mlp: 3 x (W,b)
  const int P_NE   = 77;   // node_enc
  const int P_PE   = 81;   // point_enc
  const int P_TC   = 85;   // topo_convs: 2 x 8(sorted: ef,ep,fe,ff,fp,pe,pf,pp) x (Wl,bl,Wr)
  const int P_TN   = 133;  // topo_norms: 2 x {edge,face,point} x (g,b)

  // ---- workspace bump allocator ----
  char* wsp = (char*)d_ws; size_t off = 0;
  auto alloc = [&](size_t bytes) -> void* {
    void* r = wsp + off; off += (bytes + 255) & ~(size_t)255; return r;
  };
  int*            src_idx = (int*)alloc((size_t)ECLOUD * 4);
  float*          ef4     = (float*)alloc((size_t)ECLOUD * 4 * 4);
  unsigned short* te1     = (unsigned short*)alloc((size_t)ECLOUD * HC * 2);
  unsigned short* e_attr  = (unsigned short*)alloc((size_t)ECLOUD * HC * 2);
  unsigned short* hb      = (unsigned short*)alloc((size_t)NTOT * HC * 2);
  float*          h       = (float*)alloc((size_t)NTOT * HC * 4);
  float*          agg     = (float*)alloc((size_t)NTOT * HC * 4);
  float*          t2      = (float*)alloc((size_t)NTOT * HC * 4);
  float*          xp      = (float*)alloc((size_t)SP * HT * 4);
  float*          xe      = (float*)alloc((size_t)SE * HT * 4);
  float*          xf      = (float*)alloc((size_t)SF * HT * 4);
  unsigned short* xb      = (unsigned short*)alloc((size_t)SP * HT * 2);
  float*          outp    = (float*)alloc((size_t)SP * HT * 4);
  float*          oute    = (float*)alloc((size_t)SE * HT * 4);
  float*          outf    = (float*)alloc((size_t)SF * HT * 4);
  float*          accum   = (float*)alloc((size_t)SP * HT * 4);
  float*          cnt     = (float*)alloc((size_t)SP * 4);
  unsigned short* wt_eec1 = (unsigned short*)alloc(128*128*2);
  unsigned short* wt_ne1  = (unsigned short*)alloc(128*128*2);
  unsigned short* wt_gine = (unsigned short*)alloc(12*128*128*2);   // [i]*16384
  unsigned short* wt_p1   = (unsigned short*)alloc(64*64*2);
  unsigned short* wt_e1   = (unsigned short*)alloc(64*64*2);
  unsigned short* wt_f1   = (unsigned short*)alloc(64*64*2);
  unsigned short* wt_tc   = (unsigned short*)alloc(32*64*64*2);     // [(l*8+r)*2+w]*4096
  float*          small   = (float*)alloc(64 * 1024);
  float* cloud_pool = small + 0;       // 8*128
  float* gp   = small + 1024;          // 8*64
  float* ge   = small + 1536;
  float* gf   = small + 2048;
  float* gm1  = small + 2560;
  float* gm2  = small + 3072;
  float* gm3  = small + 3584;
  float* hin  = small + 4096;          // 8*384
  float* dh1  = small + 7168;          // 8*256
  float* dh2  = small + 9216;          // 8*256

  auto tr = [&](int widx, unsigned short* dst, int Kd, int Nd) {
    int tot = Kd * Nd;
    transpose_bf16<<<(tot + 255) / 256, 256, 0, stream>>>(F(widx), dst, Kd, Nd);
  };

  // ---- 0) transpose/convert all WMMA weights ----
  tr(P_EEC + 2, wt_eec1, 128, 128);
  tr(P_NE  + 2, wt_ne1, 128, 128);
  for (int l = 0; l < 6; ++l) {
    tr(P_GINE + 4*l + 0, wt_gine + (size_t)(2*l)   * 16384, 128, 128);
    tr(P_GINE + 4*l + 2, wt_gine + (size_t)(2*l+1) * 16384, 128, 128);
  }
  tr(P_PE + 2, wt_p1, 64, 64);
  tr(P_EET + 2, wt_e1, 64, 64);
  tr(P_FE + 2, wt_f1, 64, 64);
  // relation metadata (REL order): pp fp ep pf ef ff pe fe
  const int spos[8] = {7, 4, 1, 6, 0, 3, 5, 2};        // sorted-key position
  const int rst[8]  = {0, 2, 1, 0, 1, 2, 0, 2};        // 0=point 1=edge 2=face
  const int rdt[8]  = {0, 0, 0, 2, 2, 2, 1, 1};
  const int rE[8]   = {262144,131072,131072,131072,131072,131072,131072,131072};
  for (int l = 0; l < 2; ++l)
    for (int r = 0; r < 8; ++r) {
      int pb = P_TC + l*24 + spos[r]*3;
      tr(pb + 0, wt_tc + (size_t)((l*8+r)*2 + 0) * 4096, 64, 64);  // Wl
      tr(pb + 2, wt_tc + (size_t)((l*8+r)*2 + 1) * 4096, 64, 64);  // Wr
    }

  // ---- 1) KNN + edge-encoder inputs ----
  knn_kernel<<<NTOT / 64, 64, 0, stream>>>(F(0), src_idx, ef4);

  // ---- 2) edge_enc_cloud: [4->128 relu] VALU, [128->128] WMMA -> e_attr(bf16)
  dense_small<true, true><<<ECLOUD, 128, 4*4, stream>>>(ef4, F(P_EEC), F(P_EEC+1), te1, 4, 128);
  wmma_gemm<true, true, false, false, false>
      <<<dim3(ECLOUD/16, 8), 32, 0, stream>>>(te1, nullptr, wt_eec1, F(P_EEC+3), e_attr, 128, 128);

  // ---- 3) node_enc: [7->128 relu] VALU, [128->128] WMMA -> h(f32)
  dense_small<true, true><<<NTOT, 128, 7*4, stream>>>(F(0), F(P_NE), F(P_NE+1), hb, 7, 128);
  wmma_gemm<true, false, false, false, false>
      <<<dim3(NTOT/16, 8), 32, 0, stream>>>(hb, nullptr, wt_ne1, F(P_NE+3), h, 128, 128);

  // ---- 4) 6 GINE layers ----
  for (int l = 0; l < 6; ++l) {
    gine_agg_kernel<<<NTOT, 128, 0, stream>>>(h, e_attr, src_idx, agg);
    // t1 = relu((h+agg) @ W0 + b0)   (bf16 out, fused add on A)
    wmma_gemm<false, true, true, false, true>
        <<<dim3(NTOT/16, 8), 32, 0, stream>>>(h, agg, wt_gine + (size_t)(2*l)*16384,
                                              F(P_GINE + 4*l + 1), hb, 128, 128);
    // t2 = t1 @ W1 + b1
    wmma_gemm<true, false, false, false, false>
        <<<dim3(NTOT/16, 8), 32, 0, stream>>>(hb, nullptr, wt_gine + (size_t)(2*l+1)*16384,
                                              F(P_GINE + 4*l + 3), t2, 128, 128);
    // h = LN(h + t2)
    ln_add_kernel<false><<<NTOT, 128, 2*128*4, stream>>>(h, t2, F(P_CN + 2*l), F(P_CN + 2*l + 1), 128);
  }

  // ---- 5) cloud pooling ----
  segmean_kernel<<<BATCH, 128, 0, stream>>>(h, cloud_pool, PP, 128);

  // ---- 6) topo encoders ----
  dense_small<true, true><<<SP, 64, 6*4, stream>>>(F(1), F(P_PE), F(P_PE+1), xb, 6, 64);
  wmma_gemm<true, false, false, false, false>
      <<<dim3(SP/16, 4), 32, 0, stream>>>(xb, nullptr, wt_p1, F(P_PE+3), xp, 64, 64);
  dense_small<true, true><<<SE, 64, 2*4, stream>>>(F(2), F(P_EET), F(P_EET+1), xb, 2, 64);
  wmma_gemm<true, false, false, false, false>
      <<<dim3(SE/16, 4), 32, 0, stream>>>(xb, nullptr, wt_e1, F(P_EET+3), xe, 64, 64);
  dense_small<true, true><<<SF, 64, 7*4, stream>>>(F(3), F(P_FE), F(P_FE+1), xb, 7, 64);
  wmma_gemm<true, false, false, false, false>
      <<<dim3(SF/16, 4), 32, 0, stream>>>(xb, nullptr, wt_f1, F(P_FE+3), xf, 64, 64);

  // ---- 7) 2 hetero-conv layers ----
  float* xs[3]   = {xp, xe, xf};
  float* outs[3] = {outp, oute, outf};
  const int tsz[3] = {SP, SE, SF};
  for (int l = 0; l < 2; ++l) {
    for (int t = 0; t < 3; ++t)
      zero_kernel<<<(tsz[t]*HT + 255)/256, 256, 0, stream>>>(outs[t], tsz[t]*HT);
    for (int r = 0; r < 8; ++r) {
      int st = rst[r], dt = rdt[r], E = rE[r], ds = tsz[dt];
      zero_kernel<<<(ds*HT + 255)/256, 256, 0, stream>>>(accum, ds*HT);
      zero_kernel<<<(ds + 255)/256, 256, 0, stream>>>(cnt, ds);
      scatter_add_kernel<<<E, HT, 0, stream>>>(xs[st], I(9 + r), E, accum, cnt);
      divmean_kernel<<<(ds*HT + 255)/256, 256, 0, stream>>>(accum, cnt, ds*HT);
      int pb = P_TC + l*24 + spos[r]*3;
      // out[dt] += agg @ Wl + bl
      wmma_gemm<false, false, false, true, false>
          <<<dim3(ds/16, 4), 32, 0, stream>>>(accum, nullptr,
              wt_tc + (size_t)((l*8+r)*2 + 0)*4096, F(pb+1), outs[dt], 64, 64);
      // out[dt] += x[dt] @ Wr
      wmma_gemm<false, false, false, true, false>
          <<<dim3(ds/16, 4), 32, 0, stream>>>(xs[dt], nullptr,
              wt_tc + (size_t)((l*8+r)*2 + 1)*4096, nullptr, outs[dt], 64, 64);
    }
    const int noff[3] = {4, 0, 2};   // topo_norms sorted keys: edge,face,point
    for (int t = 0; t < 3; ++t)
      ln_add_kernel<true><<<tsz[t], 64, 2*64*4, stream>>>(xs[t], outs[t],
          F(P_TN + 6*l + noff[t]), F(P_TN + 6*l + noff[t] + 1), 64);
  }

  // ---- 8) batch pooling + global MLP + head ----
  segmean_kernel<<<BATCH, 64, 0, stream>>>(xp, gp, SP/BATCH, 64);
  segmean_kernel<<<BATCH, 64, 0, stream>>>(xe, ge, SE/BATCH, 64);
  segmean_kernel<<<BATCH, 64, 0, stream>>>(xf, gf, SF/BATCH, 64);

  dense_small<true,  false><<<BATCH, 64, 14*4, stream>>>(F(4), F(P_GM),   F(P_GM+1), gm1, 14, 64);
  dense_small<true,  false><<<BATCH, 64, 64*4, stream>>>(gm1,  F(P_GM+2), F(P_GM+3), gm2, 64, 64);
  dense_small<false, false><<<BATCH, 64, 64*4, stream>>>(gm2,  F(P_GM+4), F(P_GM+5), gm3, 64, 64);

  build_head_kernel<<<BATCH, 384, 0, stream>>>(cloud_pool, gp, ge, gf, gm3, hin);

  dense_small<true,  false><<<BATCH, 256, 384*4, stream>>>(hin, F(P_DH),   F(P_DH+1), dh1, 384, 256);
  dense_small<true,  false><<<BATCH, 256, 256*4, stream>>>(dh1, F(P_DH+2), F(P_DH+3), dh2, 256, 256);
  dense_small<false, false><<<BATCH, 1,   256*4, stream>>>(dh2, F(P_DH+4), F(P_DH+5),
                                                           (float*)d_out, 256, 1);
}
